// E2EBoltConventionalTraining_63050119905629
// MI455X (gfx1250) — matrix operand order repaired and checked by reading, not compiled, not run on hardware
//
#include <hip/hip_runtime.h>

typedef __attribute__((ext_vector_type(16))) _Float16 v16h;
typedef __attribute__((ext_vector_type(8)))  float    v8f;

#define NROWS 8192   // B*S
#define HDIM  128
#define KSUB  8
#define CENT  4096
#define DSUBV 16
#define ODIM  6

// ---------------------------------------------------------------------------
// Prep: centroids -> f16, 0.5*||c||^2, LUT = c @ W3r, and a 32B f16 zero page
// used by pq_kernel's upper half-wave (K=16..31 zero padding).
// one thread per (k, c); 32768 threads total.
// ---------------------------------------------------------------------------
__global__ __launch_bounds__(256)
void prep_kernel(const float* __restrict__ centroids,   // [K,C,16]
                 const float* __restrict__ W3,          // [128,6]
                 _Float16* __restrict__ ch,             // [K,C,16] f16
                 float* __restrict__ halfnorm,          // [K,C]
                 float* __restrict__ lut,               // [K,C,6]
                 _Float16* __restrict__ zeroblk) {      // [16] f16 zeros
  int idx = blockIdx.x * blockDim.x + threadIdx.x;
  if (idx >= KSUB * CENT) return;
  if (idx < 16) zeroblk[idx] = (_Float16)0.0f;
  int k = idx >> 12;                       // idx / 4096
  const float* cp = centroids + (size_t)idx * DSUBV;
  float v[DSUBV];
  float nrm = 0.f;
#pragma unroll
  for (int d = 0; d < DSUBV; ++d) {
    v[d] = cp[d];
    nrm += v[d] * v[d];
    ch[(size_t)idx * DSUBV + d] = (_Float16)v[d];
  }
  halfnorm[idx] = 0.5f * nrm;
#pragma unroll
  for (int o = 0; o < ODIM; ++o) {
    float s = 0.f;
#pragma unroll
    for (int d = 0; d < DSUBV; ++d) s = fmaf(v[d], W3[(k * DSUBV + d) * ODIM + o], s);
    lut[(size_t)idx * ODIM + o] = s;
  }
}

// ---------------------------------------------------------------------------
// MLP: h2 = relu(relu(z@W1+b1)@W2+b2), exact f32. One 128-thread block per row.
// ---------------------------------------------------------------------------
__global__ __launch_bounds__(128)
void mlp_kernel(const float* __restrict__ z,    // [N,3]
                const float* __restrict__ W1,   // [3,128]
                const float* __restrict__ b1,   // [128]
                const float* __restrict__ W2,   // [128,128]
                const float* __restrict__ b2,   // [128]
                float* __restrict__ h2) {       // [N,128]
  __shared__ float h1s[HDIM];
  int n = blockIdx.x;
  int t = threadIdx.x;
  float z0 = z[n * 3 + 0], z1 = z[n * 3 + 1], z2 = z[n * 3 + 2];
  float a = b1[t];
  a = fmaf(z0, W1[0 * HDIM + t], a);
  a = fmaf(z1, W1[1 * HDIM + t], a);
  a = fmaf(z2, W1[2 * HDIM + t], a);
  h1s[t] = a > 0.f ? a : 0.f;
  __syncthreads();
  float acc = b2[t];
#pragma unroll 8
  for (int h = 0; h < HDIM; ++h) acc = fmaf(h1s[h], W2[h * HDIM + t], acc);
  h2[(size_t)n * HDIM + t] = acc > 0.f ? acc : 0.f;
}

// ---------------------------------------------------------------------------
// PQ encode: per (16-row tile, subspace k) wave, scan all 4096 centroids in
// 16-centroid column tiles with v_wmma_f32_16x16x32_f16 (K=16 zero-padded to
// 32 via a stride-0 zero page for lanes 16-31). 4 tiles per iteration so the
// argmin VALU of one tile fills the WMMA->VALU hazard slots of another.
// Running argmin of (0.5*||c||^2 - dot), then 16-lane shfl_xor reduce with
// lowest-index tie-break.
// ---------------------------------------------------------------------------
__global__ __launch_bounds__(32)
void pq_kernel(const float* __restrict__ h2,        // [N,128]
               const _Float16* __restrict__ ch,     // [K,C,16] f16
               const float* __restrict__ halfnorm,  // [K,C]
               const _Float16* __restrict__ zeroblk,// [16] f16 zeros
               int* __restrict__ codes) {           // [N,K]
  const int lane = threadIdx.x;     // 0..31, full wave32
  const int tile = blockIdx.x;      // 0..511 (row tile)
  const int k    = blockIdx.y;      // 0..7   (subspace)
  const int half = lane >> 4;       // 0: K=0..15 | 1: K=16..31 zero pad
  const int l16  = lane & 15;

  // A fragment: row = tile*16 + l16, lanes 0-15 elements 0..15 = x[row][k*16 .. +15],
  // lanes 16-31 = 0 (those are K=16..31 of the padded K dimension).
  v16h afrag = {};
  {
    const float* xp = h2 + (size_t)(tile * 16 + l16) * HDIM + k * DSUBV;
#pragma unroll
    for (int i = 0; i < 16; ++i) afrag[i] = half ? (_Float16)0.0f : (_Float16)xp[i];
  }

  float best[8];
  int   bcol[8];
#pragma unroll
  for (int j = 0; j < 8; ++j) { best[j] = 3.0e38f; bcol[j] = 0; }

  // B fragment source: lanes 0-15 read column `col` (K=0..15 = the centroid
  // vector); lanes 16-31 always read the zero page (K=16..31 padding).
  // Loop-invariant base + stride select -> unconditional b128 loads.
  const _Float16* cbase   = ch + (size_t)k * CENT * DSUBV;
  const _Float16* bbase   = (half == 0) ? cbase : zeroblk;
  const size_t    cstride = (half == 0) ? (size_t)DSUBV : 0;
  const float*    hbase   = halfnorm + (size_t)k * CENT;

  for (int t = 0; t < CENT / 64; ++t) {
    const int col0 = t * 64 + l16;
    const int col1 = col0 + 16;
    const int col2 = col0 + 32;
    const int col3 = col0 + 48;

    v16h b0 = *(const v16h*)(bbase + (size_t)col0 * cstride);
    v16h b1 = *(const v16h*)(bbase + (size_t)col1 * cstride);
    v16h b2 = *(const v16h*)(bbase + (size_t)col2 * cstride);
    v16h b3 = *(const v16h*)(bbase + (size_t)col3 * cstride);
    const float hn0 = hbase[col0];
    const float hn1 = hbase[col1];
    const float hn2 = hbase[col2];
    const float hn3 = hbase[col3];

    v8f acc0 = {}, acc1 = {}, acc2 = {}, acc3 = {};
    acc0 = __builtin_amdgcn_wmma_f32_16x16x32_f16(
        false, afrag, false, b0, (short)0, acc0, false, false);
    acc1 = __builtin_amdgcn_wmma_f32_16x16x32_f16(
        false, afrag, false, b1, (short)0, acc1, false, false);
    acc2 = __builtin_amdgcn_wmma_f32_16x16x32_f16(
        false, afrag, false, b2, (short)0, acc2, false, false);
    acc3 = __builtin_amdgcn_wmma_f32_16x16x32_f16(
        false, afrag, false, b3, (short)0, acc3, false, false);

    // Strictly increasing column order per lane => strict '<' keeps the
    // lowest-index minimum (matches jnp.argmin tie semantics).
#pragma unroll
    for (int j = 0; j < 8; ++j) {   // slot j <-> row (tile*16 + half*8 + j)
      float d = hn0 - acc0[j];
      if (d < best[j]) { best[j] = d; bcol[j] = col0; }
    }
#pragma unroll
    for (int j = 0; j < 8; ++j) {
      float d = hn1 - acc1[j];
      if (d < best[j]) { best[j] = d; bcol[j] = col1; }
    }
#pragma unroll
    for (int j = 0; j < 8; ++j) {
      float d = hn2 - acc2[j];
      if (d < best[j]) { best[j] = d; bcol[j] = col2; }
    }
#pragma unroll
    for (int j = 0; j < 8; ++j) {
      float d = hn3 - acc3[j];
      if (d < best[j]) { best[j] = d; bcol[j] = col3; }
    }
  }

  // Reduce across the 16 lanes of each half-wave (cols partitioned by residue).
#pragma unroll
  for (int j = 0; j < 8; ++j) {
    float s = best[j];
    int   c = bcol[j];
#pragma unroll
    for (int m = 8; m >= 1; m >>= 1) {
      float s2 = __shfl_xor(s, m, 32);
      int   c2 = __shfl_xor(c, m, 32);
      if (s2 < s || (s2 == s && c2 < c)) { s = s2; c = c2; }
    }
    if (l16 == 0) {
      int row = tile * 16 + half * 8 + j;
      codes[row * KSUB + k] = c;
    }
  }
}

// ---------------------------------------------------------------------------
// Gather: out[n,o] = b3[o] + sum_k lut[k, codes[n,k], o]
// ---------------------------------------------------------------------------
__global__ __launch_bounds__(256)
void gather_kernel(const int* __restrict__ codes,  // [N,K]
                   const float* __restrict__ lut,  // [K,C,6]
                   const float* __restrict__ b3,   // [6]
                   float* __restrict__ out) {      // [N,6]
  int n = blockIdx.x * blockDim.x + threadIdx.x;
  if (n >= NROWS) return;
  float r[ODIM];
#pragma unroll
  for (int o = 0; o < ODIM; ++o) r[o] = b3[o];
#pragma unroll
  for (int k = 0; k < KSUB; ++k) {
    int c = codes[n * KSUB + k];
    const float* lp = lut + (size_t)(k * CENT + c) * ODIM;
#pragma unroll
    for (int o = 0; o < ODIM; ++o) r[o] += lp[o];
  }
#pragma unroll
  for (int o = 0; o < ODIM; ++o) out[n * ODIM + o] = r[o];
}

// ---------------------------------------------------------------------------
extern "C" void kernel_launch(void* const* d_in, const int* in_sizes, int n_in,
                              void* d_out, int out_size, void* d_ws, size_t ws_size,
                              hipStream_t stream) {
  (void)in_sizes; (void)n_in; (void)out_size; (void)ws_size;
  const float* z         = (const float*)d_in[0];  // [32,256,3]
  const float* W1        = (const float*)d_in[1];  // [3,128]
  const float* b1        = (const float*)d_in[2];  // [128]
  const float* W2        = (const float*)d_in[3];  // [128,128]
  const float* b2        = (const float*)d_in[4];  // [128]
  const float* centroids = (const float*)d_in[5];  // [8,4096,16]
  const float* W3        = (const float*)d_in[6];  // [128,6]
  const float* b3        = (const float*)d_in[7];  // [6]

  // Workspace layout (all 256B aligned):
  char* ws = (char*)d_ws;
  float*    h2       = (float*)(ws + 0);            // 8192*128*4  = 4,194,304
  _Float16* ch       = (_Float16*)(ws + 4194304);   // 8*4096*16*2 = 1,048,576
  float*    halfnorm = (float*)(ws + 5242880);      // 8*4096*4    =   131,072
  float*    lut      = (float*)(ws + 5373952);      // 8*4096*6*4  =   786,432
  int*      codes    = (int*)(ws + 6160384);        // 8192*8*4    =   262,144
  _Float16* zeroblk  = (_Float16*)(ws + 6422528);   // 16*2 = 32 (pad to 256)

  prep_kernel<<<(KSUB * CENT + 255) / 256, 256, 0, stream>>>(centroids, W3, ch, halfnorm, lut, zeroblk);
  mlp_kernel<<<NROWS, 128, 0, stream>>>(z, W1, b1, W2, b2, h2);
  pq_kernel<<<dim3(NROWS / 16, KSUB), 32, 0, stream>>>(h2, ch, halfnorm, zeroblk, codes);
  gather_kernel<<<(NROWS + 255) / 256, 256, 0, stream>>>(codes, lut, b3, (float*)d_out);
}
